// ModelS_11742440587772
// MI455X (gfx1250) — matrix operand order repaired and checked by reading, not compiled, not run on hardware
//
#include <hip/hip_runtime.h>
#include <math.h>

// ---------------------------------------------------------------------------
// Types for CDNA5 WMMA (wave32): v_wmma_f32_16x16x32_bf16
// ---------------------------------------------------------------------------
typedef __attribute__((ext_vector_type(16))) __bf16     v16bf;
typedef __attribute__((ext_vector_type(8)))  float      v8f;
typedef __attribute__((ext_vector_type(4)))  unsigned   v4u;

union Frag16 { v16bf v; v4u q[2]; };

// CDNA5 async global->LDS copy (ASYNCcnt-tracked). The builtin takes
// (int4 addrspace(1)* gsrc, int4 addrspace(3)* ldst, imm offset, imm cpol)
// per the toolchain's declared signature.
#if __has_builtin(__builtin_amdgcn_global_load_async_to_lds_b128) && \
    __has_builtin(__builtin_amdgcn_s_wait_asynccnt)
#define USE_ASYNC_LDS 1
typedef int v4i_async __attribute__((vector_size(16)));
typedef __attribute__((address_space(1))) v4i_async as1_v4i;
typedef __attribute__((address_space(3))) v4i_async as3_v4i;
#endif

__device__ __forceinline__ v8f wmma_bf16(v16bf a, v16bf b, v8f c) {
    // 8 args: (neg_a, A, neg_b, B, c_mod, C, reuse_a, reuse_b)
    return __builtin_amdgcn_wmma_f32_16x16x32_bf16(false, a, false, b, (short)0, c, false, false);
}

__device__ __forceinline__ unsigned short f2bf(float f) {
    union { float f; unsigned u; } v; v.f = f;
    unsigned u = v.u;
    unsigned r = u + 0x7FFFu + ((u >> 16) & 1u);   // round-to-nearest-even
    return (unsigned short)(r >> 16);
}
__device__ __forceinline__ float bf2f(unsigned short h) {
    union { unsigned u; float f; } v; v.u = ((unsigned)h) << 16;
    return v.f;
}
__device__ __forceinline__ float leaky(float x) { return x > 0.f ? x : 0.01f * x; }
__device__ __forceinline__ float sigm(float x)  { return 1.f / (1.f + expf(-x)); }

// Model constants
#define BATCH 256
#define HID   300        // ENC == DEC == 300
#define GATES 1200       // 4*HID
#define ZS    10
#define KA_ENC 672       // pad(25+300 -> 352) + pad(300 -> 320)
#define KA_DEC 352       // pad(10 -> 32)      + pad(300 -> 320)
#define KT_ENC 21        // 672/32
#define KT_DEC 11        // 352/32
#define NT_G   75        // 1200/16
#define MT_G   16        // 256/16

// ---------------------------------------------------------------------------
// Weight packing: B-fragment layout.
//   packed[((nt*KT + kt)*32 + lane)*16 + e]  where
//   n = nt*16 + (lane&15),  k = kt*32 + (e>=8 ? 16 : 0) + (lane>=16 ? 8 : 0) + (e&7)
// Gate weights: A = [inA(pad PA) | inB(pad to Kpad)] -> W = [Wa | Wb]
// ---------------------------------------------------------------------------
__global__ void k_pack_gates(const float* __restrict__ Wa, int Ka, int PA,
                             const float* __restrict__ Wb, int Kb,
                             unsigned short* __restrict__ dst, int ktiles) {
    int idx = blockIdx.x * 256 + threadIdx.x;
    int total = NT_G * ktiles * 512;
    if (idx >= total) return;
    int e    = idx & 15;
    int lane = (idx >> 4) & 31;
    int kt   = (idx >> 9) % ktiles;
    int nt   = (idx >> 9) / ktiles;
    int n = nt * 16 + (lane & 15);
    int k = kt * 32 + (e >> 3) * 16 + ((lane >> 4) & 1) * 8 + (e & 7);
    float v;
    if (k < PA)  v = (k < Ka) ? Wa[n * Ka + k] : 0.f;
    else { int kk = k - PA; v = (kk < Kb) ? Wb[n * Kb + kk] : 0.f; }
    dst[idx] = f2bf(v);
}

// conv2 weights (64,16,3,3) -> implicit-GEMM B fragments, k = tap*16 + ci, Kpad=160
__global__ void k_pack_conv2(const float* __restrict__ W, unsigned short* __restrict__ dst) {
    int idx = blockIdx.x * 256 + threadIdx.x;
    if (idx >= 4 * 5 * 512) return;
    int e    = idx & 15;
    int lane = (idx >> 4) & 31;
    int kt   = (idx >> 9) % 5;
    int nt   = (idx >> 9) / 5;
    int co = nt * 16 + (lane & 15);
    int k  = kt * 32 + (e >> 3) * 16 + ((lane >> 4) & 1) * 8 + (e & 7);
    int tap = k >> 4, ci = k & 15;
    float v = 0.f;
    if (tap < 9) { int ky = tap / 3, kx = tap - ky * 3; v = W[((co * 16 + ci) * 3 + ky) * 3 + kx]; }
    dst[idx] = f2bf(v);
}

// ---------------------------------------------------------------------------
// Gates GEMM: C[256 x 1200] = A_bf16[256 x lda] * Wpacked^T
// One wave computes a 16M x 48N stripe (3 N-tiles share each A fragment).
// ---------------------------------------------------------------------------
__global__ void k_gemm_gates(const unsigned short* __restrict__ A, int lda,
                             const unsigned short* __restrict__ Bp, int ktiles,
                             float* __restrict__ C) {
    int ng = blockIdx.x, mt = blockIdx.y;          // ng in [0,25), 3 n-tiles each
    int lane = threadIdx.x, lhi = lane >> 4, ln15 = lane & 15;
    int m = mt * 16 + ln15;
    const unsigned short* arow = A + (size_t)m * lda;
    const unsigned short* bb0 = Bp + (((size_t)(ng * 3 + 0) * ktiles) << 9);
    const unsigned short* bb1 = Bp + (((size_t)(ng * 3 + 1) * ktiles) << 9);
    const unsigned short* bb2 = Bp + (((size_t)(ng * 3 + 2) * ktiles) << 9);
    v8f acc0 = {}, acc1 = {}, acc2 = {};
    for (int kt = 0; kt < ktiles; ++kt) {
        int koff = kt * 32 + lhi * 8;              // A frag: e0..7 -> k, e8..15 -> k+16
        Frag16 a;
        a.q[0] = *(const v4u*)(arow + koff);
        a.q[1] = *(const v4u*)(arow + koff + 16);
        int boff = (kt * 32 + lane) << 4;
        Frag16 b0; b0.q[0] = *(const v4u*)(bb0 + boff); b0.q[1] = *(const v4u*)(bb0 + boff + 8);
        Frag16 b1; b1.q[0] = *(const v4u*)(bb1 + boff); b1.q[1] = *(const v4u*)(bb1 + boff + 8);
        Frag16 b2; b2.q[0] = *(const v4u*)(bb2 + boff); b2.q[1] = *(const v4u*)(bb2 + boff + 8);
        __builtin_prefetch(bb0 + boff + 512, 0, 1);    // next k-tile of packed weights
        __builtin_prefetch(bb1 + boff + 512, 0, 1);
        __builtin_prefetch(bb2 + boff + 512, 0, 1);
        acc0 = wmma_bf16(a.v, b0.v, acc0);
        acc1 = wmma_bf16(a.v, b1.v, acc1);
        acc2 = wmma_bf16(a.v, b2.v, acc2);
    }
    for (int r = 0; r < 8; ++r) {
        int row = mt * 16 + r + lhi * 8;           // C frag: vgpr r -> M=r / M=8+r
        float* crow = C + (size_t)row * GATES + ln15;
        crow[(ng * 3 + 0) * 16] = acc0[r];
        crow[(ng * 3 + 1) * 16] = acc1[r];
        crow[(ng * 3 + 2) * 16] = acc2[r];
    }
}

// ---------------------------------------------------------------------------
// LSTM pointwise update; also writes bf16 copy of h into A-buffers
// ---------------------------------------------------------------------------
__global__ void k_lstm_update(const float* __restrict__ gates,
                              const float* __restrict__ bih, const float* __restrict__ bhh,
                              float* __restrict__ h, float* __restrict__ c,
                              unsigned short* __restrict__ dst1, int ld1, int off1,
                              unsigned short* __restrict__ dst2, int ld2, int off2) {
    int idx = blockIdx.x * 256 + threadIdx.x;
    if (idx >= BATCH * HID) return;
    int b = idx / HID, j = idx - b * HID;
    const float* g = gates + b * GATES;
    float iv = g[j]         + bih[j]         + bhh[j];
    float fv = g[HID + j]   + bih[HID + j]   + bhh[HID + j];
    float gg = g[2*HID + j] + bih[2*HID + j] + bhh[2*HID + j];
    float ov = g[3*HID + j] + bih[3*HID + j] + bhh[3*HID + j];
    float c2 = sigm(fv) * c[idx] + sigm(iv) * tanhf(gg);
    float hn = sigm(ov) * tanhf(c2);
    c[idx] = c2; h[idx] = hn;
    unsigned short hb = f2bf(hn);
    dst1[b * ld1 + off1 + j] = hb;
    if (dst2) dst2[b * ld2 + off2 + j] = hb;
}

// ---------------------------------------------------------------------------
// Attention read: p = h_dec@rp_W.T + rp_b, 5x5 bilinear gather from x -> A_enc[0..25)
// ---------------------------------------------------------------------------
__global__ void k_read(const float* __restrict__ h_dec, const float* __restrict__ x,
                       const float* __restrict__ rp_W, const float* __restrict__ rp_b,
                       unsigned short* __restrict__ A_enc) {
    int b = blockIdx.x, lane = threadIdx.x;
    const float* hd = h_dec + b * HID;
    float a0 = 0.f, a1 = 0.f, a2 = 0.f;
    for (int k = lane; k < HID; k += 32) {
        float hv = hd[k];
        a0 += hv * rp_W[k]; a1 += hv * rp_W[HID + k]; a2 += hv * rp_W[2*HID + k];
    }
    for (int off = 16; off; off >>= 1) {
        a0 += __shfl_xor(a0, off); a1 += __shfl_xor(a1, off); a2 += __shfl_xor(a2, off);
    }
    float p0 = a0 + rp_b[0], p1 = a1 + rp_b[1], p2 = a2 + rp_b[2];
    if (lane < 25) {
        int hh = lane / 5, ww = lane % 5;
        float gx = (ww + 0.5f) * 0.4f - 1.f;
        float gy = (hh + 0.5f) * 0.4f - 1.f;
        float u  = p0 * gx + p1, vv = p0 * gy + p2;
        float ix = ((u + 1.f) * 32.f - 1.f) * 0.5f;
        float iy = ((vv + 1.f) * 32.f - 1.f) * 0.5f;
        int x0 = (int)floorf(ix), y0 = (int)floorf(iy);
        float fx = ix - x0, fy = iy - y0;
        const float* img = x + b * 1024;
        float acc = 0.f;
        for (int dy = 0; dy < 2; ++dy)
            for (int dx = 0; dx < 2; ++dx) {
                int xi = x0 + dx, yi = y0 + dy;
                float wgt = (dx ? fx : 1.f - fx) * (dy ? fy : 1.f - fy);
                if (xi >= 0 && xi < 32 && yi >= 0 && yi < 32) acc += img[yi * 32 + xi] * wgt;
            }
        A_enc[b * KA_ENC + lane] = f2bf(acc);
    }
}

// ---------------------------------------------------------------------------
// mu/log_sigma/z: z = mu + eps*exp(ls) -> A_dec[0..10)
// ---------------------------------------------------------------------------
__global__ void k_mu_ls_z(const float* __restrict__ h_enc,
                          const float* __restrict__ mu_W, const float* __restrict__ mu_b,
                          const float* __restrict__ ls_W, const float* __restrict__ ls_b,
                          const float* __restrict__ eps, int t,
                          unsigned short* __restrict__ A_dec) {
    int b = blockIdx.x, j = threadIdx.x;
    __shared__ float s_mu[ZS], s_ls[ZS];
    if (j < 2 * ZS) {
        const float* Wr = (j < ZS) ? (mu_W + j * HID) : (ls_W + (j - ZS) * HID);
        const float* he = h_enc + b * HID;
        float acc = 0.f;
        for (int k = 0; k < HID; ++k) acc += he[k] * Wr[k];
        if (j < ZS) s_mu[j] = acc + mu_b[j]; else s_ls[j - ZS] = acc + ls_b[j - ZS];
    }
    __syncthreads();
    if (j < ZS) {
        float z = s_mu[j] + eps[(t * BATCH + b) * ZS + j] * expf(s_ls[j]);
        A_dec[b * KA_DEC + j] = f2bf(z);
    }
}

// ---------------------------------------------------------------------------
// Separable 3-D write splat: canvas += grid_sample(patch, diag-affine)
// ---------------------------------------------------------------------------
__global__ void k_write(const float* __restrict__ h_dec,
                        const float* __restrict__ w1_W, const float* __restrict__ w1_b,
                        const float* __restrict__ w2_W, const float* __restrict__ w2_b,
                        float* __restrict__ canvas) {
    int b = blockIdx.x, tid = threadIdx.x;
    __shared__ float sh[HID], s_patch[125], s_p[4];
    __shared__ float s_w0[3][32], s_w1[3][32];
    __shared__ int   s_i0[3][32], s_i1[3][32];
    for (int k = tid; k < HID; k += 128) sh[k] = h_dec[b * HID + k];
    __syncthreads();
    for (int o = tid; o < 129; o += 128) {
        const float* Wr; float bias;
        if (o < 125) { Wr = w2_W + o * HID; bias = w2_b[o]; }
        else         { Wr = w1_W + (o - 125) * HID; bias = w1_b[o - 125]; }
        float acc = bias;
        for (int k = 0; k < HID; ++k) acc += sh[k] * Wr[k];
        if (o < 125) s_patch[o] = acc; else s_p[o - 125] = acc;
    }
    __syncthreads();
    if (tid < 96) {
        int a = tid >> 5, i = tid & 31;
        float inv = 1.f / (s_p[0] + 1e-9f);       // s
        float tr  = -s_p[1 + a] * inv;            // t per axis
        float g = (i + 0.5f) * 0.0625f - 1.f;
        float u = inv * g + tr;
        float coord = ((u + 1.f) * 5.f - 1.f) * 0.5f;
        int i0 = (int)floorf(coord);
        float f = coord - i0;
        int i1 = i0 + 1;
        s_w0[a][i] = (i0 >= 0 && i0 < 5) ? (1.f - f) : 0.f;
        s_w1[a][i] = (i1 >= 0 && i1 < 5) ? f : 0.f;
        s_i0[a][i] = i0 < 0 ? 0 : (i0 > 4 ? 4 : i0);
        s_i1[a][i] = i1 < 0 ? 0 : (i1 > 4 ? 4 : i1);
    }
    __syncthreads();
    float* cb = canvas + (size_t)b * 32768;
    for (int v = tid; v < 32768; v += 128) {
        int d = v >> 10, hh = (v >> 5) & 31, ww = v & 31;
        float wx0 = s_w0[0][ww], wx1 = s_w1[0][ww]; int ix0 = s_i0[0][ww], ix1 = s_i1[0][ww];
        float wy0 = s_w0[1][hh], wy1 = s_w1[1][hh]; int iy0 = s_i0[1][hh], iy1 = s_i1[1][hh];
        float wz0 = s_w0[2][d],  wz1 = s_w1[2][d];  int iz0 = s_i0[2][d],  iz1 = s_i1[2][d];
        const float* P = s_patch;
        float r00 = wx0 * P[iz0*25 + iy0*5 + ix0] + wx1 * P[iz0*25 + iy0*5 + ix1];
        float r01 = wx0 * P[iz0*25 + iy1*5 + ix0] + wx1 * P[iz0*25 + iy1*5 + ix1];
        float r10 = wx0 * P[iz1*25 + iy0*5 + ix0] + wx1 * P[iz1*25 + iy0*5 + ix1];
        float r11 = wx0 * P[iz1*25 + iy1*5 + ix0] + wx1 * P[iz1*25 + iy1*5 + ix1];
        cb[v] += wz0 * (wy0 * r00 + wy1 * r01) + wz1 * (wy0 * r10 + wy1 * r11);
    }
}

// ---------------------------------------------------------------------------
// Rotation resample of canvas (angles from pose linear)
// ---------------------------------------------------------------------------
__global__ void k_vst(const float* __restrict__ h_dec,
                      const float* __restrict__ pose_W, const float* __restrict__ pose_b,
                      const float* __restrict__ canvas, float* __restrict__ yout) {
    int b = blockIdx.x, tid = threadIdx.x;
    __shared__ float s_a[3];
    if (tid < 3) {
        const float* hd = h_dec + b * HID;
        const float* Wr = pose_W + tid * HID;
        float acc = pose_b[tid];
        for (int k = 0; k < HID; ++k) acc += hd[k] * Wr[k];
        s_a[tid] = acc;
    }
    __syncthreads();
    float c0 = cosf(s_a[0]), c1 = cosf(s_a[1]), c2 = cosf(s_a[2]);
    float s0 = sinf(s_a[0]), s1 = sinf(s_a[1]), s2 = sinf(s_a[2]);
    float t00 =  c1*c2,            t01 = -c1*s2,            t02 =  s1;
    float t10 =  s0*s1*c2 + c0*s2, t11 = -s0*s1*s2 + c0*c2, t12 = -s0*c1;
    float t20 = -c0*s1*c2 + s0*s2, t21 =  c0*s1*s2 + s0*c2, t22 =  c0*c1;
    const float* cb = canvas + (size_t)b * 32768;
    for (int v = tid; v < 32768; v += 256) {
        int d = v >> 10, hh = (v >> 5) & 31, ww = v & 31;
        float gx = (ww + 0.5f) * 0.0625f - 1.f;
        float gy = (hh + 0.5f) * 0.0625f - 1.f;
        float gz = (d  + 0.5f) * 0.0625f - 1.f;
        float ix = ((t00*gx + t01*gy + t02*gz + 1.f) * 32.f - 1.f) * 0.5f;
        float iy = ((t10*gx + t11*gy + t12*gz + 1.f) * 32.f - 1.f) * 0.5f;
        float iz = ((t20*gx + t21*gy + t22*gz + 1.f) * 32.f - 1.f) * 0.5f;
        int x0 = (int)floorf(ix), y0 = (int)floorf(iy), z0 = (int)floorf(iz);
        float fx = ix - x0, fy = iy - y0, fz = iz - z0;
        int x1 = x0 + 1, y1 = y0 + 1, z1 = z0 + 1;
        float wx0 = (x0 >= 0 && x0 < 32) ? (1.f - fx) : 0.f;
        float wx1 = (x1 >= 0 && x1 < 32) ? fx : 0.f;
        float wy0 = (y0 >= 0 && y0 < 32) ? (1.f - fy) : 0.f;
        float wy1 = (y1 >= 0 && y1 < 32) ? fy : 0.f;
        float wz0 = (z0 >= 0 && z0 < 32) ? (1.f - fz) : 0.f;
        float wz1 = (z1 >= 0 && z1 < 32) ? fz : 0.f;
        int x0c = x0 < 0 ? 0 : (x0 > 31 ? 31 : x0), x1c = x1 < 0 ? 0 : (x1 > 31 ? 31 : x1);
        int y0c = y0 < 0 ? 0 : (y0 > 31 ? 31 : y0), y1c = y1 < 0 ? 0 : (y1 > 31 ? 31 : y1);
        int z0c = z0 < 0 ? 0 : (z0 > 31 ? 31 : z0), z1c = z1 < 0 ? 0 : (z1 > 31 ? 31 : z1);
        float r00 = wx0 * cb[z0c*1024 + y0c*32 + x0c] + wx1 * cb[z0c*1024 + y0c*32 + x1c];
        float r01 = wx0 * cb[z0c*1024 + y1c*32 + x0c] + wx1 * cb[z0c*1024 + y1c*32 + x1c];
        float r10 = wx0 * cb[z1c*1024 + y0c*32 + x0c] + wx1 * cb[z1c*1024 + y0c*32 + x1c];
        float r11 = wx0 * cb[z1c*1024 + y1c*32 + x0c] + wx1 * cb[z1c*1024 + y1c*32 + x1c];
        yout[(size_t)b * 32768 + v] = wz0 * (wy0 * r00 + wy1 * r01) + wz1 * (wy0 * r10 + wy1 * r11);
    }
}

// ---------------------------------------------------------------------------
// conv3d (16,1,32,1,1) VALID over depth + leaky -> act16 bf16 [b][h][w][ci]
// ---------------------------------------------------------------------------
__global__ void k_conv3(const float* __restrict__ y, const float* __restrict__ c3_W,
                        const float* __restrict__ c3_b, unsigned short* __restrict__ act16) {
    __shared__ float s_w[512], s_b[16];
    int tid = threadIdx.x;
    for (int k = tid; k < 512; k += 256) s_w[k] = c3_W[k];
    if (tid < 16) s_b[tid] = c3_b[tid];
    __syncthreads();
    int idx = blockIdx.x * 256 + tid;     // b*1024 + hw
    int b = idx >> 10, hw = idx & 1023;
    float acc[16];
    for (int co = 0; co < 16; ++co) acc[co] = s_b[co];
    const float* yb = y + (size_t)b * 32768 + hw;
    for (int d = 0; d < 32; ++d) {
        float v = yb[d * 1024];
        for (int co = 0; co < 16; ++co) acc[co] += v * s_w[co * 32 + d];
    }
    union { unsigned short s[16]; v4u q[2]; } o;
    for (int co = 0; co < 16; ++co) o.s[co] = f2bf(leaky(acc[co]));
    v4u* dst = (v4u*)(act16 + (size_t)idx * 16);
    dst[0] = o.q[0]; dst[1] = o.q[1];
}

// ---------------------------------------------------------------------------
// Fused conv2(16->64,3x3,pad1) WMMA implicit GEMM + leaky + conv21(64->1,3x3) + leaky
// One block per (image, 8-row band). act64 band lives in LDS (bf16).
// im2col A-fragments are built once per M-tile and reused across all 4 N-tiles.
// ---------------------------------------------------------------------------
__global__ void k_conv2_fused(const unsigned short* __restrict__ act16,
                              const unsigned short* __restrict__ Wc2,
                              const float* __restrict__ c2_b,
                              const float* __restrict__ c21_W, const float* __restrict__ c21_b,
                              float* __restrict__ xp) {
    int b = blockIdx.x, band = blockIdx.y, r0 = band * 8;
    int tid = threadIdx.x;
    __shared__ unsigned short s16[12 * 32 * 16];   // input rows r0-2 .. r0+9, chan-last
    __shared__ unsigned short s64[10 * 32 * 64];   // act64 rows r0-1 .. r0+8, chan-last
#ifdef USE_ASYNC_LDS
    {
        for (int i = tid; i < 12 * 64; i += 256) {
            int lr = i >> 6, within = i & 63;
            int gr = r0 - 2 + lr;
            if (gr >= 0 && gr < 32) {
                __builtin_amdgcn_global_load_async_to_lds_b128(
                    (as1_v4i*)(act16 + (((size_t)b * 2048 + (size_t)gr * 64 + within) << 3)),
                    (as3_v4i*)(s16 + ((size_t)i << 3)),
                    0, 0);
            } else {
                v4u z = {0u, 0u, 0u, 0u};
                ((v4u*)s16)[i] = z;
            }
        }
        __builtin_amdgcn_s_wait_asynccnt(0);
    }
#else
    {
        v4u* d4 = (v4u*)s16;
        const v4u* g4 = (const v4u*)act16;
        for (int i = tid; i < 12 * 64; i += 256) {
            int lr = i >> 6, within = i & 63;
            int gr = r0 - 2 + lr;
            v4u v = {0u, 0u, 0u, 0u};
            if (gr >= 0 && gr < 32) v = g4[(size_t)b * 2048 + gr * 64 + within];
            d4[i] = v;
        }
    }
#endif
    __syncthreads();
    int wv = tid >> 5, lane = tid & 31, lhi = lane >> 4, ln15 = lane & 15;
    for (int tm = wv; tm < 20; tm += 8) {            // 20 M-tiles over 8 waves
        int m = tm * 16 + ln15;                      // pixel index inside 10-row band
        int la = m >> 5, cc = m & 31;
        // Build all 5 K-step A fragments once (K = 160 = 9 taps*16ch, zero-padded)
        Frag16 afr[5];
        for (int kt = 0; kt < 5; ++kt)
            for (int hh = 0; hh < 2; ++hh) {
                int t = kt * 2 + hh;
                v4u u = {0u, 0u, 0u, 0u};
                if (t < 9) {
                    int ky = t / 3, kx = t - ky * 3;
                    int lr = la + ky, ic = cc + kx - 1;
                    if (ic >= 0 && ic < 32)
                        u = *(const v4u*)&s16[((lr * 32 + ic) << 4) + lhi * 8];
                }
                afr[kt].q[hh] = u;
            }
        // Reuse the A fragments across all 4 N-tiles
        for (int tn = 0; tn < 4; ++tn) {
            v8f acc = {};
            for (int kt = 0; kt < 5; ++kt) {
                const v4u* bp = (const v4u*)(Wc2 + (((tn * 5 + kt) * 32 + lane) << 4));
                Frag16 bfr; bfr.q[0] = bp[0]; bfr.q[1] = bp[1];
                acc = wmma_bf16(afr[kt].v, bfr.v, acc);
            }
            int co = tn * 16 + ln15;
            float bias = c2_b[co];
            for (int r = 0; r < 8; ++r) {
                int ms = tm * 16 + r + lhi * 8;
                int las = ms >> 5, cs = ms & 31;
                int ar = r0 - 1 + las;               // global act64 row
                float val = 0.f;
                if (ar >= 0 && ar < 32) val = leaky(acc[r] + bias);
                s64[((las * 32 + cs) << 6) + co] = f2bf(val);
            }
        }
    }
    __syncthreads();
    // conv21: one output pixel per thread (8 rows x 32 cols)
    int oh = r0 + (tid >> 5), ow = tid & 31;
    int lac = (tid >> 5) + 1;                        // local act64 row of (oh)
    float acc = c21_b[0];
    for (int ky = 0; ky < 3; ++ky) {
        int lr = lac + ky - 1;
        for (int kx = 0; kx < 3; ++kx) {
            int ic = ow + kx - 1;
            if (ic < 0 || ic >= 32) continue;
            const unsigned short* row = &s64[(lr * 32 + ic) << 6];
            for (int co = 0; co < 64; ++co)
                acc += bf2f(row[co]) * c21_W[(co * 3 + ky) * 3 + kx];
        }
    }
    xp[b * 1024 + oh * 32 + ow] = leaky(acc);
}

// ---------------------------------------------------------------------------
// Host orchestration
// ---------------------------------------------------------------------------
extern "C" void kernel_launch(void* const* d_in, const int* in_sizes, int n_in,
                              void* d_out, int out_size, void* d_ws, size_t ws_size,
                              hipStream_t stream) {
    (void)in_sizes; (void)n_in; (void)out_size; (void)ws_size;
    const float* x       = (const float*)d_in[0];
    const float* eps     = (const float*)d_in[1];
    const float* enc_Wih = (const float*)d_in[2];
    const float* enc_Whh = (const float*)d_in[3];
    const float* enc_bih = (const float*)d_in[4];
    const float* enc_bhh = (const float*)d_in[5];
    const float* dec_Wih = (const float*)d_in[6];
    const float* dec_Whh = (const float*)d_in[7];
    const float* dec_bih = (const float*)d_in[8];
    const float* dec_bhh = (const float*)d_in[9];
    const float* mu_W    = (const float*)d_in[10];
    const float* mu_b    = (const float*)d_in[11];
    const float* ls_W    = (const float*)d_in[12];
    const float* ls_b    = (const float*)d_in[13];
    const float* w1_W    = (const float*)d_in[14];
    const float* w1_b    = (const float*)d_in[15];
    const float* w2_W    = (const float*)d_in[16];
    const float* w2_b    = (const float*)d_in[17];
    const float* pose_W  = (const float*)d_in[18];
    const float* pose_b  = (const float*)d_in[19];
    const float* rp_W    = (const float*)d_in[20];
    const float* rp_b    = (const float*)d_in[21];
    const float* c3_W    = (const float*)d_in[22];
    const float* c3_b    = (const float*)d_in[23];
    const float* c2_W    = (const float*)d_in[24];
    const float* c2_b    = (const float*)d_in[25];
    const float* c21_W   = (const float*)d_in[26];
    const float* c21_b   = (const float*)d_in[27];

    float* xp     = (float*)d_out;                  // (B,1,32,32)
    float* canvas = xp + (size_t)BATCH * 1024;      // (B,32768) -- second output

    char* p = (char*)d_ws;
    auto carve = [&](size_t bytes) -> char* {
        char* r = p; p += (bytes + 255) & ~(size_t)255; return r;
    };
    float* h_enc = (float*)carve((size_t)BATCH * HID * 4);
    float* c_enc = (float*)carve((size_t)BATCH * HID * 4);
    float* h_dec = (float*)carve((size_t)BATCH * HID * 4);
    float* c_dec = (float*)carve((size_t)BATCH * HID * 4);
    float* gates = (float*)carve((size_t)BATCH * GATES * 4);
    unsigned short* A_enc = (unsigned short*)carve((size_t)BATCH * KA_ENC * 2);
    unsigned short* A_dec = (unsigned short*)carve((size_t)BATCH * KA_DEC * 2);
    unsigned short* Wenc  = (unsigned short*)carve((size_t)NT_G * KT_ENC * 512 * 2);
    unsigned short* Wdec  = (unsigned short*)carve((size_t)NT_G * KT_DEC * 512 * 2);
    unsigned short* Wc2p  = (unsigned short*)carve((size_t)4 * 5 * 512 * 2);
    float* ybuf = (float*)carve((size_t)BATCH * 32768 * 4);
    unsigned short* act16 = (unsigned short*)carve((size_t)BATCH * 1024 * 16 * 2);

    (void)hipMemsetAsync(h_enc, 0, (size_t)BATCH * HID * 4, stream);
    (void)hipMemsetAsync(c_enc, 0, (size_t)BATCH * HID * 4, stream);
    (void)hipMemsetAsync(h_dec, 0, (size_t)BATCH * HID * 4, stream);
    (void)hipMemsetAsync(c_dec, 0, (size_t)BATCH * HID * 4, stream);
    (void)hipMemsetAsync(A_enc, 0, (size_t)BATCH * KA_ENC * 2, stream);
    (void)hipMemsetAsync(A_dec, 0, (size_t)BATCH * KA_DEC * 2, stream);
    (void)hipMemsetAsync(canvas, 0, (size_t)BATCH * 32768 * 4, stream);

    k_pack_gates<<<(NT_G * KT_ENC * 512 + 255) / 256, 256, 0, stream>>>(
        enc_Wih, 325, 352, enc_Whh, 300, Wenc, KT_ENC);
    k_pack_gates<<<(NT_G * KT_DEC * 512 + 255) / 256, 256, 0, stream>>>(
        dec_Wih, 10, 32, dec_Whh, 300, Wdec, KT_DEC);
    k_pack_conv2<<<(4 * 5 * 512 + 255) / 256, 256, 0, stream>>>(c2_W, Wc2p);

    for (int t = 0; t < 10; ++t) {
        k_read<<<BATCH, 32, 0, stream>>>(h_dec, x, rp_W, rp_b, A_enc);
        k_gemm_gates<<<dim3(NT_G / 3, MT_G), 32, 0, stream>>>(A_enc, KA_ENC, Wenc, KT_ENC, gates);
        k_lstm_update<<<(BATCH * HID + 255) / 256, 256, 0, stream>>>(
            gates, enc_bih, enc_bhh, h_enc, c_enc,
            A_enc, KA_ENC, 352, (unsigned short*)nullptr, 0, 0);
        k_mu_ls_z<<<BATCH, 32, 0, stream>>>(h_enc, mu_W, mu_b, ls_W, ls_b, eps, t, A_dec);
        k_gemm_gates<<<dim3(NT_G / 3, MT_G), 32, 0, stream>>>(A_dec, KA_DEC, Wdec, KT_DEC, gates);
        k_lstm_update<<<(BATCH * HID + 255) / 256, 256, 0, stream>>>(
            gates, dec_bih, dec_bhh, h_dec, c_dec,
            A_dec, KA_DEC, 32, A_enc, KA_ENC, 25);
        k_write<<<BATCH, 128, 0, stream>>>(h_dec, w1_W, w1_b, w2_W, w2_b, canvas);
    }
    k_vst<<<BATCH, 256, 0, stream>>>(h_dec, pose_W, pose_b, canvas, ybuf);
    k_conv3<<<BATCH * 1024 / 256, 256, 0, stream>>>(ybuf, c3_W, c3_b, act16);
    k_conv2_fused<<<dim3(BATCH, 4), 256, 0, stream>>>(act16, Wc2p, c2_b, c21_W, c21_b, xp);
}
